// GNN_64106681860685
// MI455X (gfx1250) — matrix operand order repaired and checked by reading, not compiled, not run on hardware
//
#include <hip/hip_runtime.h>
#include <hip/hip_bf16.h>

#define N_NODES 100000
#define E_EDGES 1000000
#define IN_DIM  128
#define HID     256
#define G_GRAPHS 64
#define BN_EPS  1e-5f

typedef __attribute__((ext_vector_type(16))) __bf16 v16bf;
typedef __attribute__((ext_vector_type(8)))  float  v8f;

union Frag { v16bf v; uint4 q[2]; };

__device__ __forceinline__ unsigned short f2bf(float f) {
  unsigned int u = __float_as_uint(f);
  u += 0x7FFFu + ((u >> 16) & 1u);   // round-to-nearest-even
  return (unsigned short)(u >> 16);
}

// ---------- elementwise converts / packing ----------

__global__ void f32_to_bf16_k(const float* __restrict__ in,
                              unsigned short* __restrict__ out, size_t n) {
  size_t i = (size_t)blockIdx.x * 256u + threadIdx.x;
  if (i < n) out[i] = f2bf(in[i]);
}

// h_neigh = agg / max(deg,1) in bf16
__global__ void neigh_fin_k(const float* __restrict__ agg,
                            const float* __restrict__ deg,
                            unsigned short* __restrict__ out) {
  size_t i = (size_t)blockIdx.x * 256u + threadIdx.x;
  if (i < (size_t)N_NODES * IN_DIM)
    out[i] = f2bf(agg[i] / fmaxf(deg[i >> 7], 1.0f));   // IN_DIM=128
}

// Pack W[IN,HID] row-major f32 -> bf16 WMMA B-fragment layout:
// idx = ((kc*16 + nt)*32 + lane)*16 + e  holds  W[kc*32 + (lane/16)*16 + e][nt*16 + lane%16]
__global__ void pack_w_k(const float* __restrict__ W, unsigned short* __restrict__ out) {
  int idx = blockIdx.x * 256 + threadIdx.x;
  if (idx >= IN_DIM * HID) return;
  int e    = idx & 15;
  int lane = (idx >> 4) & 31;
  int t    = idx >> 9;
  int nt   = t & 15;
  int kc   = t >> 4;
  int k    = kc * 32 + (lane >> 4) * 16 + e;
  int c    = nt * 16 + (lane & 15);
  out[idx] = f2bf(W[k * HID + c]);
}

// ---------- edge aggregation (one wave per edge) ----------

__global__ __launch_bounds__(256)
void edge_agg_k(const float* __restrict__ feat, const int* __restrict__ src,
                const int* __restrict__ dst, float* __restrict__ agg,
                float* __restrict__ deg) {
  int wid  = (blockIdx.x * 256 + threadIdx.x) >> 5;
  int lane = threadIdx.x & 31;
  if (wid >= E_EDGES) return;
  int s = src[wid], d = dst[wid];
  float4 v = *(const float4*)(feat + (size_t)s * IN_DIM + lane * 4);
  float* ap = agg + (size_t)d * IN_DIM + lane * 4;
  atomicAdd(ap + 0, v.x);
  atomicAdd(ap + 1, v.y);
  atomicAdd(ap + 2, v.z);
  atomicAdd(ap + 3, v.w);
  if (lane == 0) atomicAdd(&deg[d], 1.0f);
}

__global__ void count_g_k(const int* __restrict__ gid, float* __restrict__ cnt) {
  __shared__ float sc[G_GRAPHS];
  if (threadIdx.x < G_GRAPHS) sc[threadIdx.x] = 0.f;
  __syncthreads();
  for (int i = blockIdx.x * 256 + threadIdx.x; i < N_NODES; i += gridDim.x * 256)
    atomicAdd(&sc[gid[i]], 1.0f);
  __syncthreads();
  if (threadIdx.x < G_GRAPHS) atomicAdd(&cnt[threadIdx.x], sc[threadIdx.x]);
}

// ---------- fused SAGEConv GEMM + relu + BN-stats + per-graph pooling ----------
// block = 512 threads = 16 waves; wave w handles n-tile w; grid-stride over m-tiles.
// All 16 A-fragment loads issued up-front per tile so they drain during the WMMA chain.

__global__ __launch_bounds__(512)
void gemm_conv_k(const unsigned short* __restrict__ featb,
                 const unsigned short* __restrict__ neighb,
                 const unsigned short* __restrict__ WpS,
                 const unsigned short* __restrict__ WpN,
                 const float* __restrict__ b_conv,
                 const int* __restrict__ graph_id,
                 float* __restrict__ praw,
                 float* __restrict__ colsum,
                 float* __restrict__ colsq) {
  const int lane = threadIdx.x & 31;
  const int nt   = threadIdx.x >> 5;    // 0..15
  const int kh   = lane >> 4;           // 0/1
  const int nidx = lane & 15;
  const int col  = nt * 16 + nidx;

  // Preload all B fragments (K=128 -> 4 chunks, two weight matrices): 64 VGPRs
  Frag bs[4], bn[4];
#pragma unroll
  for (int kc = 0; kc < 4; ++kc) {
    const unsigned short* ps = WpS + (((kc * 16 + nt) * 32 + lane) * 16);
    const unsigned short* pn = WpN + (((kc * 16 + nt) * 32 + lane) * 16);
    bs[kc].q[0] = *(const uint4*)(ps);
    bs[kc].q[1] = *(const uint4*)(ps + 8);
    bn[kc].q[0] = *(const uint4*)(pn);
    bn[kc].q[1] = *(const uint4*)(pn + 8);
  }

  const float bias = b_conv[col];
  float csum = 0.f, csq = 0.f;
  const int MT = N_NODES / 16;

  for (int m = blockIdx.x; m < MT; m += gridDim.x) {
    const int row = m * 16 + nidx;      // A: M = lane%16
    const unsigned short* fp = featb  + (size_t)row * IN_DIM;
    const unsigned short* np = neighb + (size_t)row * IN_DIM;

    // prefetch next grid-stride tile (wave 0 only; 1 byte/lane pulls whole lines)
    if (nt == 0) {
      int mn = m + gridDim.x;
      if (mn < MT) {
        const size_t nr = (size_t)(mn * 16 + nidx) * IN_DIM + kh * 64;
        __builtin_prefetch(featb  + nr, 0, 3);
        __builtin_prefetch(neighb + nr, 0, 3);
      }
    }

    // Issue ALL A loads for this tile (16 x b128) before the WMMA chain.
    Frag a[4], an[4];
#pragma unroll
    for (int kc = 0; kc < 4; ++kc) {
      const int kb = kc * 32 + kh * 16; // A: K = kh*16 + e (contiguous, matches packed B)
      a[kc].q[0]  = *(const uint4*)(fp + kb);
      a[kc].q[1]  = *(const uint4*)(fp + kb + 8);
      an[kc].q[0] = *(const uint4*)(np + kb);
      an[kc].q[1] = *(const uint4*)(np + kb + 8);
    }

    v8f c = {0.f, 0.f, 0.f, 0.f, 0.f, 0.f, 0.f, 0.f};
#pragma unroll
    for (int kc = 0; kc < 4; ++kc) {
      c = __builtin_amdgcn_wmma_f32_16x16x32_bf16(false, a[kc].v,  false, bs[kc].v, (short)0, c, false, false);
      c = __builtin_amdgcn_wmma_f32_16x16x32_bf16(false, an[kc].v, false, bn[kc].v, (short)0, c, false, false);
    }

    // epilogue: C/D layout -> lane holds column `col`, rows m*16 + kh*8 + r
    const int rbase = m * 16 + kh * 8;
    int g = graph_id[rbase];
    float acc = 0.f;
#pragma unroll
    for (int r = 0; r < 8; ++r) {
      float v = c[r] + bias;
      v = v > 0.f ? v : 0.f;            // relu
      csum += v; csq += v * v;          // BN stats
      int gr = graph_id[rbase + r];     // graph_id sorted -> run-length flush
      if (gr != g) { atomicAdd(&praw[g * HID + col], acc); acc = 0.f; g = gr; }
      acc += v;
    }
    atomicAdd(&praw[g * HID + col], acc);
  }
  // combine the two row-halves (lane L and L^16 share the same column)
  float osum = __shfl_xor(csum, 16, 32);
  float osq  = __shfl_xor(csq,  16, 32);
  if (kh == 0) {
    atomicAdd(&colsum[col], csum + osum);
    atomicAdd(&colsq[col],  csq + osq);
  }
}

// ---------- BN scale/shift from accumulated column stats ----------

__global__ void bn_scale_k(const float* __restrict__ colsum, const float* __restrict__ colsq,
                           const float* __restrict__ gamma, const float* __restrict__ beta,
                           float count, float* __restrict__ scale, float* __restrict__ shift) {
  int c = threadIdx.x;
  float mu  = colsum[c] / count;
  float var = colsq[c] / count - mu * mu;
  float s   = gamma[c] * rsqrtf(var + BN_EPS);
  scale[c] = s;
  shift[c] = beta[c] - mu * s;
}

// pooled[g,c] = scale[c]*sum_relu[g,c] + shift[c]*cnt[g]
__global__ void pooled_fin_k(const float* __restrict__ praw, const float* __restrict__ cnt,
                             const float* __restrict__ scale, const float* __restrict__ shift,
                             float* __restrict__ pooled) {
  int idx = blockIdx.x * 256 + threadIdx.x;
  if (idx >= G_GRAPHS * HID) return;
  int g = idx >> 8, c = idx & 255;
  pooled[idx] = scale[c] * praw[idx] + shift[c] * cnt[g];
}

// ---------- head: z = relu(x @ W_lp + b_lp), accumulate column stats ----------

__global__ __launch_bounds__(256)
void lp_matmul_k(const float* __restrict__ x, const float* __restrict__ W,
                 const float* __restrict__ b, float* __restrict__ z,
                 float* __restrict__ colsum, float* __restrict__ colsq) {
  __shared__ float xs[HID];
  int g = blockIdx.x, c = threadIdx.x;
  xs[c] = x[g * HID + c];
  __syncthreads();
  float acc = b[c];
#pragma unroll 8
  for (int k = 0; k < HID; ++k) acc = fmaf(xs[k], W[k * HID + c], acc);
  float v = fmaxf(acc, 0.f);
  z[g * HID + c] = v;
  atomicAdd(&colsum[c], v);
  atomicAdd(&colsq[c], v * v);
}

__global__ void bn_apply_k(const float* __restrict__ z, const float* __restrict__ scale,
                           const float* __restrict__ shift, float* __restrict__ out0) {
  int idx = blockIdx.x * 256 + threadIdx.x;
  if (idx >= G_GRAPHS * HID) return;
  int c = idx & 255;
  out0[idx] = z[idx] * scale[c] + shift[c];
}

__global__ __launch_bounds__(256)
void log_softmax_k(const float* __restrict__ x, float* __restrict__ out) {
  __shared__ float red[HID];
  int g = blockIdx.x, c = threadIdx.x;
  float v = x[g * HID + c];
  red[c] = v; __syncthreads();
  for (int s = 128; s > 0; s >>= 1) { if (c < s) red[c] = fmaxf(red[c], red[c + s]); __syncthreads(); }
  float mx = red[0]; __syncthreads();
  float e = expf(v - mx);
  red[c] = e; __syncthreads();
  for (int s = 128; s > 0; s >>= 1) { if (c < s) red[c] += red[c + s]; __syncthreads(); }
  float lse = logf(red[0]);
  out[g * HID + c] = v - mx - lse;
}

// ---------- host launch ----------

extern "C" void kernel_launch(void* const* d_in, const int* in_sizes, int n_in,
                              void* d_out, int out_size, void* d_ws, size_t ws_size,
                              hipStream_t stream) {
  (void)in_sizes; (void)n_in; (void)out_size; (void)ws_size;
  const float* feat       = (const float*)d_in[0];
  const float* W_self     = (const float*)d_in[1];
  const float* W_neigh    = (const float*)d_in[2];
  const float* b_conv     = (const float*)d_in[3];
  const float* gamma_conv = (const float*)d_in[4];
  const float* beta_conv  = (const float*)d_in[5];
  const float* W_lp       = (const float*)d_in[6];
  const float* b_lp       = (const float*)d_in[7];
  const float* gamma_lp   = (const float*)d_in[8];
  const float* beta_lp    = (const float*)d_in[9];
  const int*   src        = (const int*)d_in[10];
  const int*   dst        = (const int*)d_in[11];
  const int*   gid        = (const int*)d_in[12];

  float* out_ls = (float*)d_out;                       // log_softmax(out2) [64,256]
  float* out_p  = (float*)d_out + G_GRAPHS * HID;      // p [64,256]

  const size_t NI = (size_t)N_NODES * IN_DIM;
  char* w = (char*)d_ws;
  size_t off = 0;
  auto alloc = [&](size_t bytes) -> void* {
    void* p = w + off;
    off = (off + bytes + 255) & ~(size_t)255;
    return p;
  };

  unsigned short* featb  = (unsigned short*)alloc(NI * 2);
  unsigned short* neighb = (unsigned short*)alloc(NI * 2);
  float*          agg    = (float*)alloc(NI * 4);
  float*          deg    = (float*)alloc((size_t)N_NODES * 4);
  unsigned short* WpS    = (unsigned short*)alloc((size_t)IN_DIM * HID * 2);
  unsigned short* WpN    = (unsigned short*)alloc((size_t)IN_DIM * HID * 2);
  float*          praw   = (float*)alloc((size_t)G_GRAPHS * HID * 4);
  float*          pooled = (float*)alloc((size_t)G_GRAPHS * HID * 4);
  float*          z1     = (float*)alloc((size_t)G_GRAPHS * HID * 4);
  float*          z2     = (float*)alloc((size_t)G_GRAPHS * HID * 4);
  float*          cnt    = (float*)alloc((size_t)G_GRAPHS * 4);
  float*          stats  = (float*)alloc(12 * HID * 4);
  float* sA = stats + 0 * HID,  *qA = stats + 1 * HID;
  float* scA = stats + 2 * HID, *shA = stats + 3 * HID;
  float* s1 = stats + 4 * HID,  *q1 = stats + 5 * HID;
  float* sc1 = stats + 6 * HID, *sh1 = stats + 7 * HID;
  float* s2 = stats + 8 * HID,  *q2 = stats + 9 * HID;
  float* sc2 = stats + 10 * HID, *sh2 = stats + 11 * HID;

  // zero all atomic accumulators (graph-capture-legal memset nodes)
  hipMemsetAsync(agg, 0, NI * 4, stream);
  hipMemsetAsync(deg, 0, (size_t)N_NODES * 4, stream);
  hipMemsetAsync(praw, 0, (size_t)G_GRAPHS * HID * 4, stream);
  hipMemsetAsync(cnt, 0, (size_t)G_GRAPHS * 4, stream);
  hipMemsetAsync(stats, 0, 12 * HID * 4, stream);

  // precision conversion + weight packing
  f32_to_bf16_k<<<(unsigned)((NI + 255) / 256), 256, 0, stream>>>(feat, featb, NI);
  pack_w_k<<<(IN_DIM * HID + 255) / 256, 256, 0, stream>>>(W_self, WpS);
  pack_w_k<<<(IN_DIM * HID + 255) / 256, 256, 0, stream>>>(W_neigh, WpN);

  // edge aggregation (wave per edge) + degree
  edge_agg_k<<<(E_EDGES * 32) / 256, 256, 0, stream>>>(feat, src, dst, agg, deg);
  neigh_fin_k<<<(unsigned)((NI + 255) / 256), 256, 0, stream>>>(agg, deg, neighb);
  count_g_k<<<391, 256, 0, stream>>>(gid, cnt);

  // fused WMMA GEMM + relu + BN stats + per-graph pooled sums
  gemm_conv_k<<<625, 512, 0, stream>>>(featb, neighb, WpS, WpN, b_conv, gid, praw, sA, qA);
  bn_scale_k<<<1, 256, 0, stream>>>(sA, qA, gamma_conv, beta_conv, (float)N_NODES, scA, shA);
  pooled_fin_k<<<64, 256, 0, stream>>>(praw, cnt, scA, shA, pooled);

  // head: p = lp(pooled)
  lp_matmul_k<<<64, 256, 0, stream>>>(pooled, W_lp, b_lp, z1, s1, q1);
  bn_scale_k<<<1, 256, 0, stream>>>(s1, q1, gamma_lp, beta_lp, (float)G_GRAPHS, sc1, sh1);
  bn_apply_k<<<64, 256, 0, stream>>>(z1, sc1, sh1, out_p);

  // out2 = lp(p)
  lp_matmul_k<<<64, 256, 0, stream>>>(out_p, W_lp, b_lp, z2, s2, q2);
  bn_scale_k<<<1, 256, 0, stream>>>(s2, q2, gamma_lp, beta_lp, (float)G_GRAPHS, sc2, sh2);
  bn_apply_k<<<64, 256, 0, stream>>>(z2, sc2, sh2, z1);   // reuse z1 as out2

  // first tuple element: log_softmax(out2)
  log_softmax_k<<<64, 256, 0, stream>>>(z1, out_ls);
}
